// Dual_44100724196042
// MI455X (gfx1250) — matrix operand order repaired and checked by reading, not compiled, not run on hardware
//
#include <hip/hip_runtime.h>
#include <hip/hip_bf16.h>

// ---------------- problem constants ----------------
#define N_USERC 3000
#define N_ITEMC 5000
#define N_NODEC 8000
#define DDIM    64
#define EDGESC  262144
#define BATCHC  1024
#define LLEN    50
// TAU=0.2 NEG_W=0.1 PR_W=1.0 CON_W=0.001

typedef __attribute__((ext_vector_type(16))) __bf16        v16bf;
typedef __attribute__((ext_vector_type(8)))  float         v8f;
typedef __attribute__((ext_vector_type(4)))  float         f32x4;
typedef __attribute__((ext_vector_type(4)))  unsigned int  u32x4;

union ABf {
  v16bf v;
  unsigned short u[16];
  u32x4 q[2];
};

__device__ __forceinline__ unsigned short f2bf(float f) {
  unsigned int b = __float_as_uint(f);
  b += 0x7FFFu + ((b >> 16) & 1u);          // round to nearest even
  return (unsigned short)(b >> 16);
}

__device__ __forceinline__ float waveReduce32(float v) {
  v += __shfl_xor(v, 16, 32);
  v += __shfl_xor(v,  8, 32);
  v += __shfl_xor(v,  4, 32);
  v += __shfl_xor(v,  2, 32);
  v += __shfl_xor(v,  1, 32);
  return v;
}

__device__ __forceinline__ void atomicAddF(float* p, float v) {
  (void)__hip_atomic_fetch_add(p, v, __ATOMIC_RELAXED, __HIP_MEMORY_SCOPE_AGENT);
}

// ---------------- utility ----------------
__global__ void k_fzero(float* __restrict__ p, int n) {
  int i = blockIdx.x * blockDim.x + threadIdx.x;
  if (i < n) p[i] = 0.0f;
}

// Pack W = [W1 | W2] (8000 x 128) transposed to bf16: Wb[n*8000+k]
__global__ void k_prep_w(const float* __restrict__ W1, const float* __restrict__ W2,
                         unsigned short* __restrict__ Wb) {
  int idx = blockIdx.x * blockDim.x + threadIdx.x;   // 128*8000 exact
  int n = idx / N_NODEC;
  int k = idx - n * N_NODEC;
  float v = (n < DDIM) ? W1[(size_t)k * DDIM + n] : W2[(size_t)k * DDIM + (n - DDIM)];
  Wb[(size_t)n * N_NODEC + k] = f2bf(v);
}

// ---------------- GEMM1: e12 = feature(8000x8000) @ Wcat(8000x128), bf16 WMMA ----------------
// block = 128 threads = 4 waves, each wave owns one 16-row strip (grid 125 -> 500 strips).
// B tile (32 x 128 bf16 = 8 KB) is shared by all 4 waves: staged through double-buffered LDS,
// one barrier per k-step; WMMA B fragments come from ds_load_b128.
__global__ __launch_bounds__(128) void k_gemm1(const float* __restrict__ feat,
                                               const unsigned short* __restrict__ Wb,
                                               float* __restrict__ e12) {
  __shared__ u32x4 Bs[2][128][4];                    // [buf][n][16B chunk], row = 64 B
  int tid  = threadIdx.x;                            // 0..127 == B-row n
  int wv   = tid >> 5;
  int lane = tid & 31;
  int m  = lane & 15;
  int hi = lane >> 4;                                // 0/1
  int i0 = (blockIdx.x * 4 + wv) * 16;               // strip base row
  const float* arow = feat + (size_t)(i0 + m) * N_NODEC;
  const unsigned short* wrow = Wb + (size_t)tid * N_NODEC;

  v8f z = {};
  v8f c[8];
  #pragma unroll
  for (int t = 0; t < 8; ++t) c[t] = z;

  for (int k0 = 0; k0 < N_NODEC; k0 += 32) {
    int p = (k0 >> 5) & 1;
    // cooperative B tile load: thread tid brings row n=tid, 64 bytes (4 x b128)
    const u32x4* src = reinterpret_cast<const u32x4*>(wrow + k0);
    Bs[p][tid][0] = src[0];
    Bs[p][tid][1] = src[1];
    Bs[p][tid][2] = src[2];
    Bs[p][tid][3] = src[3];
    // A fragment loads (streaming, read-once): K = kb+0..7 and kb+16..23 (kb = hi*8)
    int kb = k0 + hi * 8;
    f32x4 f0 = __builtin_nontemporal_load(reinterpret_cast<const f32x4*>(arow + kb));
    f32x4 f1 = __builtin_nontemporal_load(reinterpret_cast<const f32x4*>(arow + kb + 4));
    f32x4 f2 = __builtin_nontemporal_load(reinterpret_cast<const f32x4*>(arow + kb + 16));
    f32x4 f3 = __builtin_nontemporal_load(reinterpret_cast<const f32x4*>(arow + kb + 20));
    __syncthreads();                                 // B tile visible to all 4 waves
    ABf a;
    a.u[0]=f2bf(f0.x); a.u[1]=f2bf(f0.y); a.u[2]=f2bf(f0.z); a.u[3]=f2bf(f0.w);
    a.u[4]=f2bf(f1.x); a.u[5]=f2bf(f1.y); a.u[6]=f2bf(f1.z); a.u[7]=f2bf(f1.w);
    a.u[8]=f2bf(f2.x); a.u[9]=f2bf(f2.y); a.u[10]=f2bf(f2.z); a.u[11]=f2bf(f2.w);
    a.u[12]=f2bf(f3.x); a.u[13]=f2bf(f3.y); a.u[14]=f2bf(f3.z); a.u[15]=f2bf(f3.w);
    #pragma unroll
    for (int t = 0; t < 8; ++t) {
      // B fragment: lane holds N = t*16+m, K = hi*8 + 0..7 and hi*8 + 16..23 -> 32 B from LDS
      ABf b;
      b.q[0] = Bs[p][t * 16 + m][hi];
      b.q[1] = Bs[p][t * 16 + m][hi + 2];
      c[t] = __builtin_amdgcn_wmma_f32_16x16x32_bf16(
          false, a.v, false, b.v, (short)0, c[t], false, false);
    }
  }
  int rowbase = i0 + hi * 8;
  #pragma unroll
  for (int t = 0; t < 8; ++t)
    #pragma unroll
    for (int r = 0; r < 8; ++r)
      e12[(size_t)(rowbase + r) * 128 + t * 16 + m] = c[t][r];
}

// ---------------- SPMM: dst[row] += val * src[col], D=128 ----------------
__global__ void k_spmm(const int* __restrict__ row, const int* __restrict__ col,
                       const float* __restrict__ val,
                       const float* __restrict__ src, float* __restrict__ dst) {
  long long idx = (long long)blockIdx.x * blockDim.x + threadIdx.x;
  if (idx >= (long long)EDGESC * 128) return;
  int e = (int)(idx >> 7);
  int d = (int)(idx & 127);
  float v = val[e] * src[(size_t)col[e] * 128 + d];
  atomicAddF(&dst[(size_t)row[e] * 128 + d], v);
}

// ---------------- gate pass 1: per-(node,s) score, accumulate mean numerator ----------------
__global__ void k_gate1(const float* __restrict__ z, int count,
                        const float* __restrict__ w1, const float* __restrict__ b1,
                        const float* __restrict__ w2, float* __restrict__ acc) {
  int node = blockIdx.x * blockDim.x + threadIdx.x;
  float w = 0.0f;
  if (node < count) {
    const float* zr = z + (size_t)node * 128;
    for (int e = 0; e < DDIM; ++e) {
      float h = b1[e];
      const float* wr = w1 + e * DDIM;
      #pragma unroll 16
      for (int d = 0; d < DDIM; ++d) h += zr[d] * wr[d];
      w += tanhf(h) * w2[e];
    }
  }
  w = waveReduce32(w);
  if ((threadIdx.x & 31) == 0) atomicAddF(acc, w);
}

// ---------------- beta = softmax(acc/count) for 4 gates ----------------
__global__ void k_beta(const float* __restrict__ acc, float* __restrict__ beta) {
  if (blockIdx.x != 0 || threadIdx.x != 0) return;
  const float counts[4] = {3000.0f, 5000.0f, 3000.0f, 5000.0f};
  for (int g = 0; g < 4; ++g) {
    float m0 = acc[g*3+0] / counts[g];
    float m1 = acc[g*3+1] / counts[g];
    float m2 = acc[g*3+2] / counts[g];
    float mx = fmaxf(m0, fmaxf(m1, m2));
    float e0 = __expf(m0 - mx), e1 = __expf(m1 - mx), e2 = __expf(m2 - mx);
    float s = e0 + e1 + e2;
    beta[g*3+0] = e0 / s; beta[g*3+1] = e1 / s; beta[g*3+2] = e2 / s;
  }
}

// ---------------- gate pass 2: out = sum_s beta_s * z_s ----------------
__global__ void k_gate2(const float* __restrict__ z0, const float* __restrict__ z1,
                        const float* __restrict__ z2, int count,
                        const float* __restrict__ beta, float* __restrict__ out) {
  int idx = blockIdx.x * blockDim.x + threadIdx.x;
  if (idx >= count * DDIM) return;
  int node = idx >> 6, d = idx & 63;
  size_t o = (size_t)node * 128 + d;
  out[idx] = beta[0] * z0[o] + beta[1] * z1[o] + beta[2] * z2[o];
}

// ---------------- pos_data_loss ----------------
__global__ void k_posloss(const int* __restrict__ nodes, const int* __restrict__ uiid,
                          const float* __restrict__ emb_re, const float* __restrict__ r_re,
                          float* __restrict__ loss) {
  int idx = blockIdx.x * blockDim.x + threadIdx.x;   // BATCH*L = 51200
  float part = 0.0f;
  if (idx < BATCHC * LLEN) {
    int b = idx / LLEN, l = idx - b * LLEN;
    int u = nodes[b];
    int iid = uiid[(size_t)u * LLEN + l];
    if (iid != N_ITEMC) {
      const float* hu = emb_re + (size_t)u * DDIM;
      const float* hi = emb_re + (size_t)(N_USERC + iid) * DDIM;
      float hpq = 0.0f;
      #pragma unroll 16
      for (int d = 0; d < DDIM; ++d) hpq += hu[d] * hi[d] * r_re[d];
      part = 0.9f * hpq * hpq - 2.0f * hpq;          // (1-NEG_W)=0.9
    }
  }
  part = waveReduce32(part);
  if ((threadIdx.x & 31) == 0) atomicAddF(loss, part);
}

// ---------------- all_data_loss: NEG_W * sum p1[d,d'] p2[d,d'] r[d] r[d'] ----------------
__global__ void k_allloss(const int* __restrict__ nodes, const float* __restrict__ emb_re,
                          const float* __restrict__ r_re, float* __restrict__ loss) {
  int idx = blockIdx.x * blockDim.x + threadIdx.x;   // 4096
  float term = 0.0f;
  if (idx < DDIM * DDIM) {
    int d = idx >> 6, dp = idx & 63;
    float p1 = 0.0f;
    for (int b = 0; b < BATCHC; ++b) {
      const float* h = emb_re + (size_t)nodes[b] * DDIM;
      p1 += h[d] * h[dp];
    }
    float p2 = 0.0f;                                  // h_i row N_ITEM is zeros -> skip
    const float* items = emb_re + (size_t)N_USERC * DDIM;
    for (int i = 0; i < N_ITEMC; ++i) {
      const float* h = items + (size_t)i * DDIM;
      p2 += h[d] * h[dp];
    }
    term = 0.1f * p1 * p2 * r_re[d] * r_re[dp];       // NEG_W=0.1
  }
  term = waveReduce32(term);
  if ((threadIdx.x & 31) == 0) atomicAddF(loss, term);
}

// ---------------- pr loss: PR_W * sum (hpq_pr - label)^2 ----------------
__global__ void k_prloss(const int* __restrict__ nodes, const float* __restrict__ emb_pr,
                         const float* __restrict__ r_pr, const float* __restrict__ lab,
                         float* __restrict__ loss) {
  long long idx = (long long)blockIdx.x * blockDim.x + threadIdx.x;  // 1024*5000
  float part = 0.0f;
  if (idx < (long long)BATCHC * N_ITEMC) {
    int b = (int)(idx / N_ITEMC);
    int i = (int)(idx - (long long)b * N_ITEMC);
    int u = nodes[b];
    const float* hu = emb_pr + (size_t)u * DDIM;
    const float* hi = emb_pr + (size_t)(N_USERC + i) * DDIM;
    float pred = 0.0f;
    #pragma unroll 16
    for (int d = 0; d < DDIM; ++d) pred += hu[d] * r_pr[d] * hi[d];
    float diff = pred - lab[(size_t)u * N_ITEMC + i];
    part = diff * diff;                               // PR_W = 1.0
  }
  part = waveReduce32(part);
  if ((threadIdx.x & 31) == 0) atomicAddF(loss, part);
}

// ---------------- row-normalize to bf16 (one wave per row, both embeddings) ----------------
__global__ void k_norm(const float* __restrict__ emb_re, const float* __restrict__ emb_pr,
                       unsigned short* __restrict__ er_b, unsigned short* __restrict__ ep_b) {
  int wave = blockIdx.x * (blockDim.x >> 5) + (threadIdx.x >> 5);
  if (wave >= N_NODEC) return;
  int lane = threadIdx.x & 31;
  const float* r1 = emb_re + (size_t)wave * DDIM;
  const float* r2 = emb_pr + (size_t)wave * DDIM;
  float a0 = r1[lane], a1 = r1[lane + 32];
  float b0 = r2[lane], b1 = r2[lane + 32];
  float s1 = waveReduce32(a0 * a0 + a1 * a1);
  float s2 = waveReduce32(b0 * b0 + b1 * b1);
  float i1 = rsqrtf(s1), i2 = rsqrtf(s2);
  er_b[(size_t)wave * DDIM + lane]      = f2bf(a0 * i1);
  er_b[(size_t)wave * DDIM + lane + 32] = f2bf(a1 * i1);
  ep_b[(size_t)wave * DDIM + lane]      = f2bf(b0 * i2);
  ep_b[(size_t)wave * DDIM + lane + 32] = f2bf(b1 * i2);
}

// ---------------- GEMM2: s = exp(er @ ep^T * 5), row sums of s and pos*s ----------------
__global__ void k_gemm2(const unsigned short* __restrict__ er,
                        const unsigned short* __restrict__ ep,
                        const float* __restrict__ pos,
                        float* __restrict__ tot_sum, float* __restrict__ pos_sum) {
  int wave = blockIdx.x * (blockDim.x >> 5) + (threadIdx.x >> 5);
  if (wave >= N_NODEC / 16) return;                   // wave-uniform
  int lane = threadIdx.x & 31;
  int m  = lane & 15;
  int hi = lane >> 4;
  int i0 = wave * 16;

  // A fragments for K=0..31 and K=32..63 (invariant over j)
  const unsigned short* arow = er + (size_t)(i0 + m) * DDIM;
  ABf a0, a1;
  a0.q[0] = *reinterpret_cast<const u32x4*>(arow + hi * 8);
  a0.q[1] = *reinterpret_cast<const u32x4*>(arow + hi * 8 + 16);
  a1.q[0] = *reinterpret_cast<const u32x4*>(arow + 32 + hi * 8);
  a1.q[1] = *reinterpret_cast<const u32x4*>(arow + 32 + hi * 8 + 16);

  // pos row base: lane covers rows i0+hi*8+r, col j0+m
  const float* prow = pos + (size_t)(i0 + hi * 8) * N_NODEC + m;

  float at[8], ap[8];
  #pragma unroll
  for (int r = 0; r < 8; ++r) { at[r] = 0.0f; ap[r] = 0.0f; }

  for (int j = 0; j < N_NODEC / 16; ++j) {
    int j0 = j * 16;
    // issue pos loads (streaming, read-once -> non-temporal)
    float pv[8];
    #pragma unroll
    for (int r = 0; r < 8; ++r)
      pv[r] = __builtin_nontemporal_load(prow + (size_t)r * N_NODEC + j0);
    // B fragments
    const unsigned short* brow = ep + (size_t)(j0 + m) * DDIM;
    ABf b0, b1;
    b0.q[0] = *reinterpret_cast<const u32x4*>(brow + hi * 8);
    b0.q[1] = *reinterpret_cast<const u32x4*>(brow + hi * 8 + 16);
    b1.q[0] = *reinterpret_cast<const u32x4*>(brow + 32 + hi * 8);
    b1.q[1] = *reinterpret_cast<const u32x4*>(brow + 32 + hi * 8 + 16);
    v8f c = {};
    c = __builtin_amdgcn_wmma_f32_16x16x32_bf16(false, a0.v, false, b0.v, (short)0, c, false, false);
    c = __builtin_amdgcn_wmma_f32_16x16x32_bf16(false, a1.v, false, b1.v, (short)0, c, false, false);
    #pragma unroll
    for (int r = 0; r < 8; ++r) {
      float s = __expf(c[r] * 5.0f);                  // 1/TAU = 5
      at[r] += s;
      ap[r] += pv[r] * s;
    }
  }
  // reduce across the 16 lanes of each half (xor 8..1 stays within half)
  #pragma unroll
  for (int r = 0; r < 8; ++r) {
    float t = at[r], p = ap[r];
    t += __shfl_xor(t, 8, 32); p += __shfl_xor(p, 8, 32);
    t += __shfl_xor(t, 4, 32); p += __shfl_xor(p, 4, 32);
    t += __shfl_xor(t, 2, 32); p += __shfl_xor(p, 2, 32);
    t += __shfl_xor(t, 1, 32); p += __shfl_xor(p, 1, 32);
    if (m == 0) {
      int row = i0 + hi * 8 + r;
      tot_sum[row] = t;
      pos_sum[row] = p;
    }
  }
}

// ---------------- contrastive: loss += CON_W * -sum log(pos/(tot-pos)) ----------------
__global__ void k_con(const float* __restrict__ tot, const float* __restrict__ ps,
                      float* __restrict__ loss) {
  int idx = blockIdx.x * blockDim.x + threadIdx.x;
  float part = 0.0f;
  if (idx < N_NODEC) {
    float p = ps[idx], t = tot[idx];
    part = -0.001f * __logf(p / (t - p));             // CON_W = 0.001
  }
  part = waveReduce32(part);
  if ((threadIdx.x & 31) == 0) atomicAddF(loss, part);
}

__global__ void k_final(const float* __restrict__ loss, float* __restrict__ out) {
  if (blockIdx.x == 0 && threadIdx.x == 0) out[0] = loss[0];
}

// ---------------- workspace layout (float offsets) ----------------
static constexpr size_t OFF_E12  = 0;        // 8000*128
static constexpr size_t OFF_TMP  = 1024000;  // 8000*128
static constexpr size_t OFF_IG   = 2048000;  // 8000*128  [i1|i2]
static constexpr size_t OFF_IM   = 3072000;  // 8000*128  [i4|i3]
static constexpr size_t OFF_ERE  = 4096000;  // 8000*64  emb_re (unnormalized)
static constexpr size_t OFF_EPR  = 4608000;  // 8000*64  emb_pr
static constexpr size_t OFF_TOT  = 5120000;  // 8000
static constexpr size_t OFF_POS  = 5128000;  // 8000
static constexpr size_t OFF_GACC = 5136000;  // 12
static constexpr size_t OFF_LOSS = 5136012;  // 1
static constexpr size_t OFF_BETA = 5136016;  // 12
static constexpr size_t OFF_WB   = 5136032;  // ushort[128*8000] -> 512000 floats
static constexpr size_t OFF_ERB  = 5648032;  // ushort[8000*64]  -> 256000 floats
static constexpr size_t OFF_EPB  = 5904032;  // ushort[8000*64]  -> 256000 floats
// total: 6,160,032 floats (~24.6 MB)

extern "C" void kernel_launch(void* const* d_in, const int* in_sizes, int n_in,
                              void* d_out, int out_size, void* d_ws, size_t ws_size,
                              hipStream_t stream) {
  const float* feature = (const float*)d_in[0];
  const float* W1      = (const float*)d_in[1];
  const float* W2      = (const float*)d_in[2];
  const float* gval    = (const float*)d_in[3];
  const float* mval    = (const float*)d_in[4];
  const float* gw1[4]  = {(const float*)d_in[5],  (const float*)d_in[8],
                          (const float*)d_in[11], (const float*)d_in[14]};
  const float* gb1[4]  = {(const float*)d_in[6],  (const float*)d_in[9],
                          (const float*)d_in[12], (const float*)d_in[15]};
  const float* gw2[4]  = {(const float*)d_in[7],  (const float*)d_in[10],
                          (const float*)d_in[13], (const float*)d_in[16]};
  const float* r_re    = (const float*)d_in[17];
  const float* r_pr    = (const float*)d_in[18];
  const float* lab     = (const float*)d_in[19];
  const float* pos     = (const float*)d_in[20];
  const int*   grow    = (const int*)d_in[21];
  const int*   gcol    = (const int*)d_in[22];
  const int*   mrow    = (const int*)d_in[23];
  const int*   mcol    = (const int*)d_in[24];
  const int*   nodes   = (const int*)d_in[25];
  const int*   uiid    = (const int*)d_in[26];

  float* ws   = (float*)d_ws;
  float* e12  = ws + OFF_E12;
  float* tmp  = ws + OFF_TMP;
  float* ig   = ws + OFF_IG;
  float* im   = ws + OFF_IM;
  float* ere  = ws + OFF_ERE;
  float* epr  = ws + OFF_EPR;
  float* tot  = ws + OFF_TOT;
  float* psum = ws + OFF_POS;
  float* gacc = ws + OFF_GACC;
  float* loss = ws + OFF_LOSS;
  float* beta = ws + OFF_BETA;
  unsigned short* Wb  = (unsigned short*)(ws + OFF_WB);
  unsigned short* erb = (unsigned short*)(ws + OFF_ERB);
  unsigned short* epb = (unsigned short*)(ws + OFF_EPB);

  // zero accumulators (gacc[12] + loss[1] contiguous)
  k_fzero<<<1, 32, 0, stream>>>(gacc, 13);

  // ---- big GEMM: e12 = feature @ [W1|W2] ----
  k_prep_w<<<4000, 256, 0, stream>>>(W1, W2, Wb);
  k_gemm1<<<125, 128, 0, stream>>>(feature, Wb, e12);

  // ---- 2-layer propagation on both graphs ----
  const int spmmBlocks = (EDGESC * 128) / 256;       // 131072
  k_fzero<<<4000, 256, 0, stream>>>(tmp, 1024000);
  k_spmm<<<spmmBlocks, 256, 0, stream>>>(grow, gcol, gval, e12, tmp);
  k_fzero<<<4000, 256, 0, stream>>>(ig, 1024000);
  k_spmm<<<spmmBlocks, 256, 0, stream>>>(grow, gcol, gval, tmp, ig);
  k_fzero<<<4000, 256, 0, stream>>>(tmp, 1024000);
  k_spmm<<<spmmBlocks, 256, 0, stream>>>(mrow, mcol, mval, e12, tmp);
  k_fzero<<<4000, 256, 0, stream>>>(im, 1024000);
  k_spmm<<<spmmBlocks, 256, 0, stream>>>(mrow, mcol, mval, tmp, im);

  // ---- gates ----
  const size_t IOFF = (size_t)N_USERC * 128;         // item row offset
  const float* zg[4][3] = {
    { ig,              ig + 64,          im + 64        },   // ure: u1,u2,u3
    { ig + IOFF,       ig + IOFF + 64,   im + IOFF + 64 },   // ire: it1,it2,it3
    { ig + 64,         im + 64,          im             },   // upr: u2,u3,u4
    { ig + IOFF + 64,  im + IOFF + 64,   im + IOFF      },   // ipr: it2,it3,it4
  };
  const int gcount[4] = {N_USERC, N_ITEMC, N_USERC, N_ITEMC};
  for (int g = 0; g < 4; ++g)
    for (int s = 0; s < 3; ++s)
      k_gate1<<<(gcount[g] + 63) / 64, 64, 0, stream>>>(
          zg[g][s], gcount[g], gw1[g], gb1[g], gw2[g], gacc + g * 3 + s);
  k_beta<<<1, 1, 0, stream>>>(gacc, beta);

  float* gout[4] = { ere, ere + (size_t)N_USERC * DDIM,
                     epr, epr + (size_t)N_USERC * DDIM };
  for (int g = 0; g < 4; ++g)
    k_gate2<<<(gcount[g] * DDIM + 255) / 256, 256, 0, stream>>>(
        zg[g][0], zg[g][1], zg[g][2], gcount[g], beta + g * 3, gout[g]);

  // ---- BPR + all-data + pr losses (unnormalized embeddings) ----
  k_posloss<<<(BATCHC * LLEN + 255) / 256, 256, 0, stream>>>(nodes, uiid, ere, r_re, loss);
  k_allloss<<<16, 256, 0, stream>>>(nodes, ere, r_re, loss);
  k_prloss<<<(BATCHC * N_ITEMC) / 256, 256, 0, stream>>>(nodes, epr, r_pr, lab, loss);

  // ---- contrastive: normalize -> bf16 -> 8000x8000 WMMA GEMM with exp/pos ----
  k_norm<<<1000, 256, 0, stream>>>(ere, epr, erb, epb);
  k_gemm2<<<63, 256, 0, stream>>>(erb, epb, pos, tot, psum);
  k_con<<<(N_NODEC + 255) / 256, 256, 0, stream>>>(tot, psum, loss);

  k_final<<<1, 1, 0, stream>>>(loss, (float*)d_out);
}
